// HierarchicalCrossAttentionLayer_11630771437633
// MI455X (gfx1250) — compile-verified, matrix operand order checked by reading, not run on hardware
//
#include <hip/hip_runtime.h>
#include <hip/hip_bf16.h>

// ---------------------------------------------------------------------------
// HierarchicalCrossAttentionLayer for MI455X (gfx1250), wave32, bf16 WMMA.
// D=256, T0=8, nQ=100, B=8, M=8192.  Rows are flattened (n, t) with n=q*8+j.
// ---------------------------------------------------------------------------

typedef __attribute__((ext_vector_type(16))) __bf16 v16bf;
typedef __attribute__((ext_vector_type(8)))  float  v8f;

#define D_C    256
#define NMEM   8192
#define SCALEF 0.0625f     // 256^-0.5
#define EPSA   1e-8f
#define LNEPS  1e-5f

enum { MODE_Q = 0, MODE_KV = 1, MODE_LN = 2, MODE_RELU = 3, MODE_TANH = 4 };

union FragU { v16bf v; unsigned int d[8]; unsigned short us[16]; };

__device__ __forceinline__ unsigned short f2bf(float f) {
  unsigned int u = __float_as_uint(f);
  u += 0x7fffu + ((u >> 16) & 1u);          // round-to-nearest-even
  return (unsigned short)(u >> 16);
}

// f32 -> bf16 (packed pairs), n must be even
__global__ void cvt_bf16_kernel(const float* __restrict__ in, unsigned short* __restrict__ out, int n2)
{
  int idx = blockIdx.x * 256 + threadIdx.x;   // over n/2 pairs
  if (idx >= n2) return;
  float2 f = ((const float2*)in)[idx];
  ((unsigned int*)out)[idx] = (unsigned int)f2bf(f.x) | ((unsigned int)f2bf(f.y) << 16);
}

// ---------------------------------------------------------------------------
// Phase 0: Hungarian matching per batch j (serial algorithm, parallel cost).
// cost[a][b] = (Cm.T)[a][b] = 1 - rn[b].cn[a]
// ---------------------------------------------------------------------------
__global__ __launch_bounds__(256) void perms_kernel(
    const float* __restrict__ tgt, int* __restrict__ perm,
    int* __restrict__ inv, float* __restrict__ last)
{
  __shared__ float cost[100 * 100];
  __shared__ float rlen[100], clen[100];
  __shared__ float hu[101], hv[101], hmin[101];
  __shared__ int   hp[101], hway[101];
  __shared__ unsigned char hused[101];
  __shared__ int   aa[100];

  const int j   = blockIdx.x;
  const int tid = threadIdx.x;
  float* lastj = last + (size_t)j * 100 * 256;

  for (int q = tid; q < 100; q += 256) perm[q * 8 + j] = q;   // frame 0 = identity
  for (int e = tid; e < 100 * 256; e += 256) {
    int q = e >> 8, c = e & 255;
    lastj[e] = tgt[(((size_t)q) * 8 + j) * 256 + c];
  }
  __syncthreads();

  for (int fr = 1; fr < 8; fr++) {
    const float* cur = tgt + (((size_t)fr * 100) * 8 + j) * 256;  // row a at +a*8*256
    for (int n = tid; n < 200; n += 256) {
      const float* vp = (n < 100) ? (lastj + (size_t)n * 256)
                                  : (cur + (size_t)(n - 100) * 8 * 256);
      float s = 0.f;
      for (int c = 0; c < 256; c++) { float v = vp[c]; s += v * v; }
      if (n < 100) rlen[n] = sqrtf(s) + 1e-6f; else clen[n - 100] = sqrtf(s) + 1e-6f;
    }
    __syncthreads();
    for (int e = tid; e < 10000; e += 256) {
      int a = e / 100, b = e % 100;
      const float* lb = lastj + (size_t)b * 256;
      const float* ca = cur + (size_t)a * 8 * 256;
      float dot = 0.f;
      for (int c = 0; c < 256; c++) dot += lb[c] * ca[c];
      float vf = 1.0f - dot / (rlen[b] * clen[a]);
      cost[a * 100 + b] = (vf == vf) ? vf : 0.0f;  // nan_to_num
    }
    __syncthreads();
    if (tid == 0) {
      const float INF = 1e18f;
      for (int k = 0; k <= 100; k++) { hu[k] = 0.f; hv[k] = 0.f; hp[k] = 0; hway[k] = 0; }
      for (int i = 1; i <= 100; i++) {
        hp[0] = i; int j0 = 0;
        for (int k = 0; k <= 100; k++) { hmin[k] = INF; hused[k] = 0; }
        while (true) {
          hused[j0] = 1;
          int i0 = hp[j0]; float delta = INF; int j1 = 0;
          for (int jj = 1; jj <= 100; jj++) {
            if (!hused[jj]) {
              float c = cost[(i0 - 1) * 100 + (jj - 1)] - hu[i0] - hv[jj];
              if (c < hmin[jj]) { hmin[jj] = c; hway[jj] = j0; }
              if (hmin[jj] < delta) { delta = hmin[jj]; j1 = jj; }
            }
          }
          for (int jj = 0; jj <= 100; jj++) {
            if (hused[jj]) { hu[hp[jj]] += delta; hv[jj] -= delta; }
            else           hmin[jj] -= delta;
          }
          j0 = j1;
          if (hp[j0] == 0) break;
        }
        while (j0) { int jn = hway[j0]; hp[j0] = hp[jn]; j0 = jn; }
      }
      for (int jj = 1; jj <= 100; jj++) aa[hp[jj] - 1] = jj - 1;
    }
    __syncthreads();
    for (int q = tid; q < 100; q += 256) perm[((size_t)fr * 100 + q) * 8 + j] = aa[q];
    for (int e = tid; e < 100 * 256; e += 256) {
      int q = e >> 8, c = e & 255;
      lastj[e] = cur[(size_t)aa[q] * 8 * 256 + c];   // last = cur[idx]
    }
    __syncthreads();
  }
  for (int e = tid; e < 800; e += 256) {
    int fr = e / 100, q = e % 100;
    int pq = perm[((size_t)fr * 100 + q) * 8 + j];
    inv[((size_t)fr * 100 + pq) * 8 + j] = q;
  }
}

// x[(q*8+j)*8 + i][c] = tgt[i][perm[i,q,j]][j][c]
__global__ void build_x_kernel(const float* __restrict__ tgt, const int* __restrict__ perm,
                               float* __restrict__ x)
{
  int idx = blockIdx.x * 256 + threadIdx.x;  // 1,638,400
  int c = idx & 255; int r = idx >> 8;
  int i = r & 7; int p = r >> 3; int j = p & 7; int q = p >> 3;
  int pq = perm[(i * 100 + q) * 8 + j];
  x[idx] = tgt[(((size_t)i * 100 + pq) * 8 + j) * 256 + c];
}

// ---------------------------------------------------------------------------
// Generic 16-row x 256-col WMMA GEMM: Y = X @ W^T (+bias)(+res -> LN)(act).
// One wave per 16-row tile, 4 waves per block. W is pre-converted bf16.
// Fully unrolled so acc[] stays in VGPRs; min-waves=1 for a big VGPR budget.
// ---------------------------------------------------------------------------
template <int MODE>
__global__ __launch_bounds__(128, 1) void proj_kernel(
    const float* __restrict__ X, const unsigned short* __restrict__ Wb,
    const float* __restrict__ bias,
    const float* __restrict__ res, const float* __restrict__ lng, const float* __restrict__ lnb,
    float* __restrict__ outf, unsigned short* __restrict__ outb, int M)
{
  const int wave = threadIdx.x >> 5;
  const int lane = threadIdx.x & 31;
  const int tile = blockIdx.x * 4 + wave;
  if (tile * 16 >= M) return;
  const int ln16 = lane & 15, g = lane >> 4;
  const int r0 = tile * 16;

  // A fragments: row = r0+ln16, slot j -> channel kc*32 + j + 8g + (j>=8 ? 8 : 0)
  v16bf afrag[8];
  {
    const float* xr = X + (size_t)(r0 + ln16) * D_C;
    #pragma unroll
    for (int kc = 0; kc < 8; kc++) {
      FragU t;
      #pragma unroll
      for (int p = 0; p < 4; p++) {
        float2 f = *(const float2*)(xr + kc * 32 + 2 * p + 8 * g);
        t.us[2 * p] = f2bf(f.x); t.us[2 * p + 1] = f2bf(f.y);
      }
      #pragma unroll
      for (int p = 0; p < 4; p++) {
        float2 f = *(const float2*)(xr + kc * 32 + 16 + 2 * p + 8 * g);
        t.us[8 + 2 * p] = f2bf(f.x); t.us[9 + 2 * p] = f2bf(f.y);
      }
      afrag[kc] = t.v;
    }
  }

  v8f acc[16];
  #pragma unroll
  for (int ct = 0; ct < 16; ct++) { v8f z = {0,0,0,0,0,0,0,0}; acc[ct] = z; }

  #pragma unroll
  for (int ct = 0; ct < 16; ct++) {
    const unsigned short* wr = Wb + (size_t)(ct * 16 + ln16) * D_C;  // B[k][n] = W[c0+n][k]
    #pragma unroll
    for (int kc = 0; kc < 8; kc++) {
      FragU b;
      #pragma unroll
      for (int p = 0; p < 4; p++)
        b.d[p]     = *(const unsigned int*)(wr + kc * 32 + 2 * p + 8 * g);
      #pragma unroll
      for (int p = 0; p < 4; p++)
        b.d[4 + p] = *(const unsigned int*)(wr + kc * 32 + 16 + 2 * p + 8 * g);
      acc[ct] = __builtin_amdgcn_wmma_f32_16x16x32_bf16(
          false, afrag[kc], false, b.v, (short)0, acc[ct], false, false);
    }
  }

  if (MODE == MODE_Q || MODE == MODE_KV || MODE == MODE_LN) {
    #pragma unroll
    for (int ct = 0; ct < 16; ct++) {
      float bc = bias[ct * 16 + ln16];
      #pragma unroll
      for (int r = 0; r < 8; r++) acc[ct][r] += bc;
    }
  }

  if (MODE == MODE_LN) {
    #pragma unroll
    for (int ct = 0; ct < 16; ct++)
      #pragma unroll
      for (int r = 0; r < 8; r++)
        acc[ct][r] += res[(size_t)(r0 + r + 8 * g) * D_C + ct * 16 + ln16];

    float mean[8], rstd[8];
    #pragma unroll
    for (int r = 0; r < 8; r++) {
      float s = 0.f, q = 0.f;
      #pragma unroll
      for (int ct = 0; ct < 16; ct++) { float v = acc[ct][r]; s += v; q += v * v; }
      #pragma unroll
      for (int off = 1; off < 16; off <<= 1) { s += __shfl_xor(s, off, 32); q += __shfl_xor(q, off, 32); }
      float m = s * (1.0f / 256.0f);
      mean[r] = m;
      rstd[r] = rsqrtf(q * (1.0f / 256.0f) - m * m + LNEPS);
    }
    #pragma unroll
    for (int ct = 0; ct < 16; ct++) {
      float gw = lng[ct * 16 + ln16], bw = lnb[ct * 16 + ln16];
      #pragma unroll
      for (int r = 0; r < 8; r++)
        outf[(size_t)(r0 + r + 8 * g) * D_C + ct * 16 + ln16] =
            (acc[ct][r] - mean[r]) * rstd[r] * gw + bw;
    }
    return;
  }

  if (MODE == MODE_RELU) {
    #pragma unroll
    for (int ct = 0; ct < 16; ct++)
      #pragma unroll
      for (int r = 0; r < 8; r++) acc[ct][r] = fmaxf(acc[ct][r], 0.0f);
  }
  if (MODE == MODE_TANH) {
    #pragma unroll
    for (int ct = 0; ct < 16; ct++)
      #pragma unroll
      for (int r = 0; r < 8; r++) acc[ct][r] = tanhf(acc[ct][r]);
  }

  #pragma unroll
  for (int ct = 0; ct < 16; ct++)
    #pragma unroll
    for (int r = 0; r < 8; r++) {
      size_t idx = (size_t)(r0 + r + 8 * g) * D_C + ct * 16 + ln16;
      if (MODE == MODE_KV) outb[idx] = f2bf(acc[ct][r]);
      else                 outf[idx] = acc[ct][r];
    }
}

// ---------------------------------------------------------------------------
// Streaming cross-attention, softmax over the frame axis t (lane-local).
// 4 waves/block share K/V LDS tiles; each wave owns one 16-(n,t)-row tile.
// ---------------------------------------------------------------------------
template <int T>
__global__ __launch_bounds__(128, 1) void attn_kernel(
    const float* __restrict__ H, const unsigned short* __restrict__ Kb,
    const unsigned short* __restrict__ Vb, float* __restrict__ O)
{
  __shared__ unsigned short Klds[32 * 256];   // 16 KB: K rows m0..m0+31, row-major
  __shared__ unsigned short VT[256 * 32];     // 16 KB: V transposed [c][m]
  __shared__ unsigned short Sst[4][16 * 32];  //  4 KB: per-wave attn staging

  const int tid = threadIdx.x;
  const int wave = tid >> 5, lane = tid & 31;
  const int ln16 = lane & 15, g = lane >> 4;
  const int r0 = (blockIdx.x * 4 + wave) * 16;

  // Q (=h) A-fragments for the whole kernel
  v16bf afrag[8];
  {
    const float* hr = H + (size_t)(r0 + ln16) * D_C;
    #pragma unroll
    for (int kc = 0; kc < 8; kc++) {
      FragU t;
      #pragma unroll
      for (int p = 0; p < 4; p++) {
        float2 f = *(const float2*)(hr + kc * 32 + 2 * p + 8 * g);
        t.us[2 * p] = f2bf(f.x); t.us[2 * p + 1] = f2bf(f.y);
      }
      #pragma unroll
      for (int p = 0; p < 4; p++) {
        float2 f = *(const float2*)(hr + kc * 32 + 16 + 2 * p + 8 * g);
        t.us[8 + 2 * p] = f2bf(f.x); t.us[9 + 2 * p] = f2bf(f.y);
      }
      afrag[kc] = t.v;
    }
  }

  v8f acc[16];
  #pragma unroll
  for (int ct = 0; ct < 16; ct++) { v8f z = {0,0,0,0,0,0,0,0}; acc[ct] = z; }
  float wsum[8] = {0.f,0.f,0.f,0.f,0.f,0.f,0.f,0.f};

  for (int m0 = 0; m0 < NMEM; m0 += 32) {
    // cooperative staging (all 128 threads)
    {
      const uint4* ks = (const uint4*)(Kb + (size_t)m0 * D_C);
      uint4* kd = (uint4*)Klds;
      for (int e = tid; e < 1024; e += 128) kd[e] = ks[e];
      const unsigned int* vs = (const unsigned int*)(Vb + (size_t)m0 * D_C);
      for (int e = tid; e < 4096; e += 128) {
        unsigned int d = vs[e];
        int r = e >> 7;             // 128 dwords per 256-ch row
        int c = (e & 127) * 2;
        VT[c * 32 + r]       = (unsigned short)(d & 0xffffu);
        VT[(c + 1) * 32 + r] = (unsigned short)(d >> 16);
      }
      if (m0 + 32 < NMEM) {        // L2 prefetch of next chunk
        __builtin_prefetch(Kb + (size_t)(m0 + 32) * D_C + tid * 64, 0, 1);
        __builtin_prefetch(Vb + (size_t)(m0 + 32) * D_C + tid * 64, 0, 1);
      }
    }
    __syncthreads();

    // S = q . k^T for 2 halves of 16 memory rows; softmax over t per lane
    #pragma unroll
    for (int half = 0; half < 2; half++) {
      v8f s = {0,0,0,0,0,0,0,0};
      #pragma unroll
      for (int kc = 0; kc < 8; kc++) {
        FragU b;
        const unsigned short* kr = Klds + (half * 16 + ln16) * D_C + kc * 32;
        #pragma unroll
        for (int p = 0; p < 4; p++) b.d[p]     = *(const unsigned int*)(kr + 2 * p + 8 * g);
        #pragma unroll
        for (int p = 0; p < 4; p++) b.d[4 + p] = *(const unsigned int*)(kr + 16 + 2 * p + 8 * g);
        s = __builtin_amdgcn_wmma_f32_16x16x32_bf16(
            false, afrag[kc], false, b.v, (short)0, s, false, false);
      }
      float er[8];
      #pragma unroll
      for (int grp = 0; grp < 8 / T; grp++) {
        float mx = -1e30f;
        #pragma unroll
        for (int r = 0; r < T; r++) mx = fmaxf(mx, s[grp * T + r] * SCALEF);
        float sum = 0.f;
        #pragma unroll
        for (int r = 0; r < T; r++) { float e = __expf(s[grp * T + r] * SCALEF - mx); er[grp * T + r] = e; sum += e; }
        float is = 1.0f / sum;
        #pragma unroll
        for (int r = 0; r < T; r++) { er[grp * T + r] *= is; wsum[grp * T + r] += er[grp * T + r]; }
      }
      unsigned short* sd = &Sst[wave][0];
      #pragma unroll
      for (int r = 0; r < 8; r++)
        sd[(r + 8 * g) * 32 + half * 16 + ln16] = f2bf(er[r]);
    }
    asm volatile("s_wait_dscnt 0" ::: "memory");

    // attn A-fragment (16x32, K = memory rows of this step)
    FragU af;
    {
      const unsigned short* sr = &Sst[wave][ln16 * 32];
      #pragma unroll
      for (int p = 0; p < 4; p++) af.d[p]     = *(const unsigned int*)(sr + 2 * p + 8 * g);
      #pragma unroll
      for (int p = 0; p < 4; p++) af.d[4 + p] = *(const unsigned int*)(sr + 16 + 2 * p + 8 * g);
    }
    // O += attn @ V
    #pragma unroll
    for (int ct = 0; ct < 16; ct++) {
      FragU b;
      const unsigned short* vr = VT + (ct * 16 + ln16) * 32;
      #pragma unroll
      for (int p = 0; p < 4; p++) b.d[p]     = *(const unsigned int*)(vr + 2 * p + 8 * g);
      #pragma unroll
      for (int p = 0; p < 4; p++) b.d[4 + p] = *(const unsigned int*)(vr + 16 + 2 * p + 8 * g);
      acc[ct] = __builtin_amdgcn_wmma_f32_16x16x32_bf16(
          false, af.v, false, b.v, (short)0, acc[ct], false, false);
    }
    __syncthreads();
  }

  // w = sum_m attn  (reduce over the 16 column-lanes of each half-group)
  #pragma unroll
  for (int r = 0; r < 8; r++) {
    float w = wsum[r];
    #pragma unroll
    for (int off = 1; off < 16; off <<= 1) w += __shfl_xor(w, off, 32);
    wsum[r] = w;
  }
  #pragma unroll
  for (int ct = 0; ct < 16; ct++)
    #pragma unroll
    for (int r = 0; r < 8; r++)
      O[(size_t)(r0 + r + 8 * g) * D_C + ct * 16 + ln16] = acc[ct][r] / (wsum[r] + EPSA);
}

// h_out[n][tt][c] = 0.5*(h_in[n][2tt][c] + h_in[n][2tt+1][c])
__global__ void pairavg_kernel(const float* __restrict__ in, float* __restrict__ out, int thalf)
{
  int idx = blockIdx.x * 256 + threadIdx.x;
  if (idx >= 800 * thalf * 256) return;
  int c = idx & 255; int r = idx >> 8;
  int tt = r % thalf; int n = r / thalf;
  out[idx] = 0.5f * (in[(size_t)(n * 2 * thalf + 2 * tt) * 256 + c] +
                     in[(size_t)(n * 2 * thalf + 2 * tt + 1) * 256 + c]);
}

// out[i][q][j][c] = x[p'][i][c] + gate[p'][c]*h[p'][c],  p' = inv[i,q,j]*8 + j
__global__ void final_kernel(const float* __restrict__ x, const float* __restrict__ gate,
                             const float* __restrict__ h, const int* __restrict__ inv,
                             float* __restrict__ out)
{
  int idx = blockIdx.x * 256 + threadIdx.x;  // 1,638,400
  int c = idx & 255; int r = idx >> 8;
  int j = r & 7; int r2 = r >> 3; int q = r2 % 100; int i = r2 / 100;
  int p = inv[(i * 100 + q) * 8 + j] * 8 + j;
  out[idx] = x[((size_t)p * 8 + i) * 256 + c] +
             gate[(size_t)p * 256 + c] * h[(size_t)p * 256 + c];
}

// ---------------------------------------------------------------------------
extern "C" void kernel_launch(void* const* d_in, const int* in_sizes, int n_in,
                              void* d_out, int out_size, void* d_ws, size_t ws_size,
                              hipStream_t stream)
{
  (void)in_sizes; (void)n_in; (void)out_size; (void)ws_size;
  const float* tgt    = (const float*)d_in[0];
  const float* memory = (const float*)d_in[1];
  const float* Wq  = (const float*)d_in[2];  const float* bq  = (const float*)d_in[3];
  const float* Wk  = (const float*)d_in[4];  const float* bk  = (const float*)d_in[5];
  const float* Wv  = (const float*)d_in[6];  const float* bv  = (const float*)d_in[7];
  const float* Wo1 = (const float*)d_in[8];  const float* bo1 = (const float*)d_in[9];
  const float* g1  = (const float*)d_in[10]; const float* be1 = (const float*)d_in[11];
  const float* Wo2 = (const float*)d_in[12]; const float* bo2 = (const float*)d_in[13];
  const float* g2  = (const float*)d_in[14]; const float* be2 = (const float*)d_in[15];
  const float* Wo3 = (const float*)d_in[16]; const float* bo3 = (const float*)d_in[17];
  const float* g3  = (const float*)d_in[18]; const float* be3 = (const float*)d_in[19];
  const float* Wg1 = (const float*)d_in[20]; const float* Wg2 = (const float*)d_in[21];

  char* w = (char*)d_ws;
  auto carve = [&](size_t bytes) -> void* {
    void* p = (void*)w; w += (bytes + 255) & ~(size_t)255; return p;
  };
  float* x    = (float*)carve((size_t)1638400 * 4);
  float* hA   = (float*)carve((size_t)1638400 * 4);
  float* hB   = (float*)carve((size_t)1638400 * 4);
  float* obuf = (float*)carve((size_t)1638400 * 4);
  unsigned short* kbf = (unsigned short*)carve((size_t)8192 * 256 * 2);
  unsigned short* vbf = (unsigned short*)carve((size_t)8192 * 256 * 2);
  float* last = (float*)carve((size_t)8 * 100 * 256 * 4);
  int*   perm = (int*)carve((size_t)6400 * 4);
  int*   inv  = (int*)carve((size_t)6400 * 4);
  float* mid  = (float*)carve((size_t)800 * 256 * 4);
  float* gate = (float*)carve((size_t)800 * 256 * 4);
  // bf16 copies of the eight 256x256 weight matrices
  unsigned short* Wbf[8];
  for (int i = 0; i < 8; i++) Wbf[i] = (unsigned short*)carve((size_t)65536 * 2);

  const float* Wsrc[8] = { Wq, Wk, Wv, Wo1, Wo2, Wo3, Wg1, Wg2 };
  for (int i = 0; i < 8; i++)
    cvt_bf16_kernel<<<128, 256, 0, stream>>>(Wsrc[i], Wbf[i], 32768);

  perms_kernel<<<8, 256, 0, stream>>>(tgt, perm, inv, last);
  build_x_kernel<<<6400, 256, 0, stream>>>(tgt, perm, x);

  proj_kernel<MODE_Q ><<<100, 128, 0, stream>>>(x,      Wbf[0], bq, nullptr, nullptr, nullptr, hA,      nullptr, 6400);
  proj_kernel<MODE_KV><<<128, 128, 0, stream>>>(memory, Wbf[1], bk, nullptr, nullptr, nullptr, nullptr, kbf,     8192);
  proj_kernel<MODE_KV><<<128, 128, 0, stream>>>(memory, Wbf[2], bv, nullptr, nullptr, nullptr, nullptr, vbf,     8192);

  // layer 1 (t = 8)
  attn_kernel<8><<<100, 128, 0, stream>>>(hA, kbf, vbf, obuf);
  proj_kernel<MODE_LN><<<100, 128, 0, stream>>>(obuf, Wbf[3], bo1, hA, g1, be1, hB, nullptr, 6400);
  pairavg_kernel<<<3200, 256, 0, stream>>>(hB, hA, 4);
  // layer 2 (t = 4)
  attn_kernel<4><<<50, 128, 0, stream>>>(hA, kbf, vbf, obuf);
  proj_kernel<MODE_LN><<<50, 128, 0, stream>>>(obuf, Wbf[4], bo2, hA, g2, be2, hB, nullptr, 3200);
  pairavg_kernel<<<1600, 256, 0, stream>>>(hB, hA, 2);
  // layer 3 (t = 2)
  attn_kernel<2><<<25, 128, 0, stream>>>(hA, kbf, vbf, obuf);
  proj_kernel<MODE_LN><<<25, 128, 0, stream>>>(obuf, Wbf[5], bo3, hA, g3, be3, hB, nullptr, 1600);
  pairavg_kernel<<<800, 256, 0, stream>>>(hB, hA, 1);

  // gate = tanh(relu(h @ Wg1^T) @ Wg2^T)
  proj_kernel<MODE_RELU><<<13, 128, 0, stream>>>(hA,  Wbf[6], nullptr, nullptr, nullptr, nullptr, mid,  nullptr, 800);
  proj_kernel<MODE_TANH><<<13, 128, 0, stream>>>(mid, Wbf[7], nullptr, nullptr, nullptr, nullptr, gate, nullptr, 800);

  final_kernel<<<6400, 256, 0, stream>>>(x, gate, hA, inv, (float*)d_out);
}
